// AdditiveUniAttention_16587163697971
// MI455X (gfx1250) — matrix operand order repaired and checked by reading, not compiled, not run on hardware
//
#include <hip/hip_runtime.h>
#include <hip/hip_bf16.h>
#include <stdint.h>

// Problem shape (fixed by the reference): B=8, M=64, L=2048, H=768, A=64.
#define BATCH 8
#define MTOK  64
#define LTOK  2048
#define HDIM  768
#define ADIM  64
#define LN_EPS 1e-5f

#define USE_ASYNC_LDS 1
#define USE_TDM       1

typedef __attribute__((ext_vector_type(16))) __bf16    v16bf;
typedef __attribute__((ext_vector_type(8)))  float     v8f;
typedef __attribute__((ext_vector_type(4)))  int       v4i;
typedef __attribute__((ext_vector_type(4)))  unsigned  v4u;
typedef __attribute__((ext_vector_type(8)))  unsigned  v8u;

union FragBF { v16bf bf; v4i i4[2]; };

__device__ __forceinline__ unsigned short f2bf(float f) {
  unsigned u = __float_as_uint(f);
  u += 0x7FFFu + ((u >> 16) & 1u);   // round-to-nearest-even bf16
  return (unsigned short)(u >> 16);
}
__device__ __forceinline__ unsigned pack2bf(float a, float b) {
  return (unsigned)f2bf(a) | ((unsigned)f2bf(b) << 16);
}
__device__ __forceinline__ float fast_tanh(float x) {
  float cx = fminf(15.0f, fmaxf(-15.0f, x));
  float e2 = __builtin_amdgcn_exp2f(cx * 2.8853900817779268f); // 2*log2(e)
  return (e2 - 1.0f) * __builtin_amdgcn_rcpf(e2 + 1.0f);
}
__device__ __forceinline__ float wred_max(float v) {
#pragma unroll
  for (int o = 16; o; o >>= 1) v = fmaxf(v, __shfl_xor(v, o, 32));
  return v;
}
__device__ __forceinline__ float wred_sum(float v) {
#pragma unroll
  for (int o = 16; o; o >>= 1) v += __shfl_xor(v, o, 32);
  return v;
}

// --- CDNA5 async global->LDS copy (ASYNCcnt) ------------------------------
__device__ __forceinline__ void async_copy_b64(unsigned lds_off, const void* g) {
#if USE_ASYNC_LDS
  unsigned long long ga = (unsigned long long)(uintptr_t)g;
  asm volatile("global_load_async_to_lds_b64 %0, %1, off"
               :: "v"(lds_off), "v"(ga) : "memory");
#endif
}
template <int N>
__device__ __forceinline__ void wait_async_le() {
#if USE_ASYNC_LDS
  asm volatile("s_wait_asynccnt %0" :: "i"(N) : "memory");
#endif
}

// --- CDNA5 Tensor Data Mover: 2D tile load, 2-byte elements (TENSORcnt) ---
__device__ __forceinline__ void tdm_load_tile_2d_b16(
    unsigned lds_addr, const void* gaddr,
    unsigned tile_w, unsigned tile_h,
    unsigned row_stride, unsigned tensor_w, unsigned tensor_h) {
#if USE_TDM
  unsigned long long ga = (unsigned long long)(uintptr_t)gaddr;
  v4u g0;
  g0[0] = 1u;                                   // count=1, user desc
  g0[1] = lds_addr;                             // lds_addr (bytes)
  g0[2] = (unsigned)ga;                         // global_addr[31:0]
  g0[3] = ((unsigned)(ga >> 32) & 0x01FFFFFFu)  // global_addr[56:32]
        | (2u << 30);                           // type = 2 ("image")
  v8u g1;
  g1[0] = (1u << 16);                           // wg_mask=0, data_size=2B
  g1[1] = (tensor_w & 0xFFFFu) << 16;
  g1[2] = ((tensor_w >> 16) & 0xFFFFu) | ((tensor_h & 0xFFFFu) << 16);
  g1[3] = ((tensor_h >> 16) & 0xFFFFu) | ((tile_w & 0xFFFFu) << 16);
  g1[4] = (tile_h & 0xFFFFu);
  g1[5] = row_stride;
  g1[6] = 0u;
  g1[7] = 0u;
  asm volatile("tensor_load_to_lds %0, %1" :: "s"(g0), "s"(g1) : "memory");
#endif
}

// ---------------------------------------------------------------------------
// out[r,c] = sum_k X[r,k]*W[c,k] + bias[c], K = HDIM = 768.
// 128x64 block tile, K-chunk 32, 8 waves. Register-relay double buffering:
// f32 global loads of chunk i+1 overlap the WMMAs of chunk i (LOADcnt),
// one barrier per K-step. BF16_OUT stores TRANSPOSED per batch [B][c][l]
// as packed b128 (C layout gives each lane 8 consecutive M).
// ---------------------------------------------------------------------------
template <bool BF16_OUT>
__global__ __launch_bounds__(256)
void wmma_gemm_xwT(const float* __restrict__ X, const float* __restrict__ W,
                   const float* __restrict__ bias, void* __restrict__ out,
                   int ldo) {
  __shared__ __align__(16) unsigned int XsU[2][128 * 16]; // [buf][row][k-pair]
  __shared__ __align__(16) unsigned int WsU[2][64 * 16];  // [buf][col][k-pair]
  const int tid  = threadIdx.x;
  const int lane = tid & 31, wv = tid >> 5;
  const int lg   = lane >> 4, lh = lane & 15;
  const int r0 = blockIdx.x * 128;
  const int c0 = blockIdx.y * 64;

  v8f acc[4] = {};
  float4 xr[4], wr[2];

  auto load_regs = [&](int kc) {
#pragma unroll
    for (int i = 0; i < 4; i++) {
      int q = i * 256 + tid; int row = q >> 3, kf = (q & 7) * 4;
      xr[i] = *(const float4*)(X + (size_t)(r0 + row) * HDIM + kc + kf);
    }
#pragma unroll
    for (int i = 0; i < 2; i++) {
      int q = i * 256 + tid; int col = q >> 3, kf = (q & 7) * 4;
      wr[i] = *(const float4*)(W + (size_t)(c0 + col) * HDIM + kc + kf);
    }
  };
  auto store_lds = [&](int buf) {
#pragma unroll
    for (int i = 0; i < 4; i++) {
      int q = i * 256 + tid; int row = q >> 3, kf = (q & 7) * 4;
      uint2 p; p.x = pack2bf(xr[i].x, xr[i].y); p.y = pack2bf(xr[i].z, xr[i].w);
      *(uint2*)(&XsU[buf][row * 16 + (kf >> 1)]) = p;
    }
#pragma unroll
    for (int i = 0; i < 2; i++) {
      int q = i * 256 + tid; int col = q >> 3, kf = (q & 7) * 4;
      uint2 p; p.x = pack2bf(wr[i].x, wr[i].y); p.y = pack2bf(wr[i].z, wr[i].w);
      *(uint2*)(&WsU[buf][col * 16 + (kf >> 1)]) = p;
    }
  };

  load_regs(0);
  store_lds(0);
  __syncthreads();

  const int NC = HDIM / 32;                       // 24 K-steps
  for (int c = 0; c < NC; ++c) {
    const int cur = c & 1;
    if (c + 1 < NC) load_regs((c + 1) * 32);      // overlaps WMMAs below

    FragBF a;   // A 16x32: v0..3 K=2v+8lg, v4..7 +16
    {
      const unsigned* base = &XsU[cur][(wv * 16 + lh) * 16 + lg * 4];
      a.i4[0] = *(const v4i*)(base);
      a.i4[1] = *(const v4i*)(base + 8);
    }
#pragma unroll
    for (int nt = 0; nt < 4; nt++) {
      const unsigned* base = &WsU[cur][(nt * 16 + lh) * 16 + lg * 4];
      FragBF bfr;                                 // B 32x16: lane=N, v0..3 K=2v+8lg, v4..7 +16
      bfr.i4[0] = *(const v4i*)(base);
      bfr.i4[1] = *(const v4i*)(base + 8);
      acc[nt] = __builtin_amdgcn_wmma_f32_16x16x32_bf16(
          false, a.bf, false, bfr.bf, (short)0, acc[nt], false, false);
    }
    if (c + 1 < NC) {
      store_lds(cur ^ 1);                         // other buffer: no hazard
      __syncthreads();
    }
  }

  if (BF16_OUT) {
    int mbase = r0 + wv * 16 + 8 * lg;            // 8 consecutive GEMM rows
    int bb = mbase >> 11, ll = mbase & (LTOK - 1);
#pragma unroll
    for (int nt = 0; nt < 4; nt++) {
      int n = c0 + nt * 16 + lh;
      float bn = bias[n];
      v4i pk;
#pragma unroll
      for (int j = 0; j < 4; j++)
        pk[j] = (int)pack2bf(acc[nt][2 * j] + bn, acc[nt][2 * j + 1] + bn);
      *(v4i*)((unsigned short*)out + ((size_t)(bb * HDIM + n)) * LTOK + ll) = pk;
    }
  } else {
#pragma unroll
    for (int nt = 0; nt < 4; nt++) {
      int n = c0 + nt * 16 + lh;
      float bn = bias[n];
#pragma unroll
      for (int v = 0; v < 8; v++) {
        int m = r0 + wv * 16 + v + 8 * lg;
        ((float*)out)[(size_t)m * ldo + n] = acc[nt][v] + bn;
      }
    }
  }
}

// ---------------------------------------------------------------------------
// scores[b,m,l] = v_b + sum_a v_w[a]*tanh(Qe[b,m,a] + Ke[b,l,a]); masked.
// ---------------------------------------------------------------------------
__global__ __launch_bounds__(256)
void scores_kernel(const float* __restrict__ Qe, const float* __restrict__ Ke,
                   const int* __restrict__ mask, const float* __restrict__ v_w,
                   const float* __restrict__ v_b, float* __restrict__ scores) {
  __shared__ float Qs[MTOK * ADIM];   // 16 KB
  __shared__ float Vs[ADIM];
  const int b = blockIdx.x, l0 = blockIdx.y * 256;
  const int tid = threadIdx.x;
  for (int i = tid; i < MTOK * ADIM; i += 256)
    Qs[i] = Qe[(size_t)b * MTOK * ADIM + i];
  if (tid < ADIM) Vs[tid] = v_w[tid];
  __syncthreads();

  const int l = l0 + tid;
  float4 ke4[16];
  const float4* kr = (const float4*)(Ke + ((size_t)b * LTOK + l) * ADIM);
#pragma unroll
  for (int i = 0; i < 16; i++) ke4[i] = kr[i];
  float4 vr[16];
#pragma unroll
  for (int i = 0; i < 16; i++) vr[i] = ((const float4*)Vs)[i];

  const bool  valid = (mask[(size_t)b * LTOK + l] == 1);
  const float vb = v_b[0];
  float* srow = scores + (size_t)(b * MTOK) * LTOK + l;
  for (int m = 0; m < MTOK; m++) {
    const float4* q4 = (const float4*)(Qs + m * ADIM);
    float s = vb;
#pragma unroll
    for (int i = 0; i < 16; i++) {
      float4 q = q4[i];
      s += vr[i].x * fast_tanh(q.x + ke4[i].x);
      s += vr[i].y * fast_tanh(q.y + ke4[i].y);
      s += vr[i].z * fast_tanh(q.z + ke4[i].z);
      s += vr[i].w * fast_tanh(q.w + ke4[i].w);
    }
    srow[(size_t)m * LTOK] = valid ? s : -3.0e38f;
  }
}

// ---------------------------------------------------------------------------
// Row softmax over L=2048 -> bf16 attn. Wave32 shuffle reductions.
// ---------------------------------------------------------------------------
__global__ __launch_bounds__(256)
void softmax_kernel(const float* __restrict__ scores,
                    unsigned short* __restrict__ attn) {
  __shared__ float red[8];
  const int row = blockIdx.x;               // b*M + m
  const int tid = threadIdx.x, lane = tid & 31, wv = tid >> 5;
  const float* s = scores + (size_t)row * LTOK;
  float v[8];
  float mx = -3.4e38f;
#pragma unroll
  for (int i = 0; i < 8; i++) { v[i] = s[tid + 256 * i]; mx = fmaxf(mx, v[i]); }
  mx = wred_max(mx);
  if (lane == 0) red[wv] = mx;
  __syncthreads();
#pragma unroll
  for (int i = 0; i < 8; i++) mx = fmaxf(mx, red[i]);
  float sum = 0.f;
#pragma unroll
  for (int i = 0; i < 8; i++) {
    v[i] = __builtin_amdgcn_exp2f((v[i] - mx) * 1.4426950408889634f);
    sum += v[i];
  }
  sum = wred_sum(sum);
  __syncthreads();
  if (lane == 0) red[wv] = sum;
  __syncthreads();
  sum = 0.f;
#pragma unroll
  for (int i = 0; i < 8; i++) sum += red[i];
  float inv = __builtin_amdgcn_rcpf(sum);
  unsigned short* arow = attn + (size_t)row * LTOK;
#pragma unroll
  for (int i = 0; i < 8; i++) arow[tid + 256 * i] = f2bf(v[i] * inv);
}

// ---------------------------------------------------------------------------
// context[b,:,n0:n0+64] = attn[b] (64xL bf16) x Vproj^T[b] (HxL bf16, [n][l]),
// then x = meta + context. Double-buffered: TDM loads the A tile (TENSORcnt),
// async b64 DMA loads the B tile (ASYNCcnt); chunk i+1 is issued before the
// threshold waits for chunk i (asynccnt<=2 / tensorcnt<=1).
// ---------------------------------------------------------------------------
__global__ __launch_bounds__(256)
void context_kernel(const unsigned short* __restrict__ attn,   // [B,M,L]
                    const unsigned short* __restrict__ vprojT, // [B,H,L]
                    const float* __restrict__ meta,            // [B,M,H]
                    float* __restrict__ xout) {                // [B,M,H]
  __shared__ __align__(16) unsigned short As[2][64 * 32];  // attn   [m][k]
  __shared__ __align__(16) unsigned short Bs[2][64 * 32];  // vprojT [n][k]
  const int b = blockIdx.x, n0 = blockIdx.y * 64;
  const int tid = threadIdx.x, lane = tid & 31, wv = tid >> 5;
  const int lg = lane >> 4, lh = lane & 15;
  const int mt = wv & 3, nc = wv >> 2;
  v8f acc[2] = {};

  auto issue = [&](int chunk, int buf) {
    const int l0 = chunk * 32;
    if (wv == 0) {
      tdm_load_tile_2d_b16((unsigned)(uintptr_t)(void*)&As[buf][0],
                           attn + (size_t)(b * MTOK) * LTOK + l0,
                           /*tile_w=*/32, /*tile_h=*/64,
                           /*row_stride=*/LTOK, /*tensor_w=*/LTOK,
                           /*tensor_h=*/MTOK);
    }
    const unsigned bs_base = (unsigned)(uintptr_t)(void*)&Bs[buf][0];
#pragma unroll
    for (int i = 0; i < 2; i++) {
      int q = i * 256 + tid;
      int n = q >> 3, kq = q & 7;
      const unsigned short* src =
          vprojT + ((size_t)(b * HDIM + n0 + n)) * LTOK + l0 + kq * 4;
      async_copy_b64(bs_base + (unsigned)(n * 64 + kq * 8), src);
    }
  };

  issue(0, 0);
  const int NC = LTOK / 32;                       // 64 K-steps
  for (int c = 0; c < NC; ++c) {
    const int cur = c & 1;
    if (c + 1 < NC) {
      issue(c + 1, cur ^ 1);                      // DMA overlaps compute
      wait_async_le<2>();                         // chunk c complete
      if (wv == 0) __builtin_amdgcn_s_wait_tensorcnt(1);
    } else {
      wait_async_le<0>();
      if (wv == 0) __builtin_amdgcn_s_wait_tensorcnt(0);
    }
    __syncthreads();

    FragBF a;
    {
      const unsigned short* ab = &As[cur][(mt * 16 + lh) * 32 + lg * 8];
      a.i4[0] = *(const v4i*)(ab);
      a.i4[1] = *(const v4i*)(ab + 16);
    }
#pragma unroll
    for (int t = 0; t < 2; t++) {
      const unsigned short* bb = &Bs[cur][((nc * 2 + t) * 16 + lh) * 32 + lg * 8];
      FragBF bfr;
      bfr.i4[0] = *(const v4i*)(bb);
      bfr.i4[1] = *(const v4i*)(bb + 16);
      acc[t] = __builtin_amdgcn_wmma_f32_16x16x32_bf16(
          false, a.bf, false, bfr.bf, (short)0, acc[t], false, false);
    }
    __syncthreads();                              // buffer reusable next iter
  }

#pragma unroll
  for (int t = 0; t < 2; t++) {
    int n = n0 + (nc * 2 + t) * 16 + lh;
#pragma unroll
    for (int v = 0; v < 8; v++) {
      int m = mt * 16 + v + 8 * lg;
      size_t idx = ((size_t)(b * MTOK + m)) * HDIM + n;
      xout[idx] = acc[t][v] + meta[idx];
    }
  }
}

// ---------------------------------------------------------------------------
// Row LayerNorm over H=768. Wave32 shuffle reductions.
// ---------------------------------------------------------------------------
__global__ __launch_bounds__(256)
void layernorm_kernel(const float* __restrict__ x, const float* __restrict__ g,
                      const float* __restrict__ be, float* __restrict__ out) {
  __shared__ float rs[8], rq[8];
  const int row = blockIdx.x, tid = threadIdx.x;
  const int lane = tid & 31, wv = tid >> 5;
  const float* xr = x + (size_t)row * HDIM;
  float v[3]; float s = 0.f, q = 0.f;
#pragma unroll
  for (int i = 0; i < 3; i++) { v[i] = xr[tid + 256 * i]; s += v[i]; q += v[i] * v[i]; }
  s = wred_sum(s); q = wred_sum(q);
  if (lane == 0) { rs[wv] = s; rq[wv] = q; }
  __syncthreads();
  s = 0.f; q = 0.f;
#pragma unroll
  for (int i = 0; i < 8; i++) { s += rs[i]; q += rq[i]; }
  float mu  = s * (1.0f / HDIM);
  float var = q * (1.0f / HDIM) - mu * mu;
  float inv = rsqrtf(var + LN_EPS);
  float* orow = out + (size_t)row * HDIM;
#pragma unroll
  for (int i = 0; i < 3; i++) {
    int c = tid + 256 * i;
    orow[c] = (v[i] - mu) * inv * g[c] + be[c];
  }
}

// ---------------------------------------------------------------------------
extern "C" void kernel_launch(void* const* d_in, const int* in_sizes, int n_in,
                              void* d_out, int out_size, void* d_ws, size_t ws_size,
                              hipStream_t stream) {
  (void)in_sizes; (void)n_in; (void)out_size; (void)ws_size;
  const float* meta = (const float*)d_in[0];
  const float* text = (const float*)d_in[1];
  const int*   mask = (const int*)d_in[2];
  const float* Wq_w = (const float*)d_in[3];
  const float* Wq_b = (const float*)d_in[4];
  const float* Wk_w = (const float*)d_in[5];
  const float* Wk_b = (const float*)d_in[6];
  const float* v_w  = (const float*)d_in[7];
  const float* v_b  = (const float*)d_in[8];
  const float* Vv_w = (const float*)d_in[9];
  const float* Vv_b = (const float*)d_in[10];
  const float* ln_g = (const float*)d_in[11];
  const float* ln_b = (const float*)d_in[12];

  // Workspace layout (bytes), 37.3 MB total.
  char* ws = (char*)d_ws;
  float*          Qe = (float*)(ws + 0);                 //  512*64*4   = 128 KB
  float*          Ke = (float*)(ws + 131072);            // 16384*64*4  = 4 MB
  unsigned short* Vp = (unsigned short*)(ws + 4325376);  // [B][H][L] bf16 24 MB
  float*          Sc = (float*)(ws + 29491200);          //  512*2048*4 = 4 MB
  unsigned short* At = (unsigned short*)(ws + 33685504); //  512*2048*2 = 2 MB
  float*          Xb = (float*)(ws + 35782656);          //  512*768*4  = 1.5 MB

  // 1) Qe = meta . Wq^T + b   [512 x 64]
  wmma_gemm_xwT<false><<<dim3(4, 1), 256, 0, stream>>>(meta, Wq_w, Wq_b, (void*)Qe, ADIM);
  // 2) Ke = text . Wk^T + b   [16384 x 64]
  wmma_gemm_xwT<false><<<dim3(128, 1), 256, 0, stream>>>(text, Wk_w, Wk_b, (void*)Ke, ADIM);
  // 3) Vproj^T = (text . Vv^T + b)^T per batch, bf16 [B][H][L]
  wmma_gemm_xwT<true><<<dim3(128, 12), 256, 0, stream>>>(text, Vv_w, Vv_b, (void*)Vp, 0);
  // 4) additive scores with mask (native exp2/rcp tanh)
  scores_kernel<<<dim3(BATCH, LTOK / 256), 256, 0, stream>>>(Qe, Ke, mask, v_w, v_b, Sc);
  // 5) softmax rows -> bf16 attn
  softmax_kernel<<<BATCH * MTOK, 256, 0, stream>>>(Sc, At);
  // 6) context GEMM (double-buffered TDM + async-LDS staging) + residual
  context_kernel<<<dim3(BATCH, HDIM / 64), 256, 0, stream>>>(At, Vp, meta, Xb);
  // 7) LayerNorm -> output
  layernorm_kernel<<<BATCH * MTOK, 256, 0, stream>>>(Xb, ln_g, ln_b, (float*)d_out);
}
